// decoderLSTM_66494683677306
// MI455X (gfx1250) — matrix operand order repaired and checked by reading, not compiled, not run on hardware
//
#include <hip/hip_runtime.h>
#include <hip/hip_bf16.h>

// ---------------- problem constants ----------------
#define B_   128
#define T_   32
#define KATT 49
#define E_   512
#define H_   512
#define VOC  10000
#define KIN  1536          // 2E + H   (fused gate-GEMM K)
#define NG   2560          // 4H + H   (i,f,g,o + sentinel gate)
#define NT   4             // N-tiles per wave (register blocking)
#define NW   8             // waves per block in GEMM kernels

typedef _Float16 half8  __attribute__((ext_vector_type(8)));
typedef _Float16 v16h   __attribute__((ext_vector_type(16)));
typedef float    v8f    __attribute__((ext_vector_type(8)));
typedef int      v4i    __attribute__((ext_vector_type(4)));

typedef __attribute__((address_space(1))) v4i* gv4ip;   // global int4*
typedef __attribute__((address_space(3))) v4i* lv4ip;   // LDS int4*

// ---- optional CDNA5 async global->LDS copy (ASYNCcnt path), guarded ----
#if defined(__has_builtin)
#  if __has_builtin(__builtin_amdgcn_global_load_async_to_lds_b128)
#    define HAVE_ASYNC_LDS 1
#  endif
#endif
#ifndef HAVE_ASYNC_LDS
#  define HAVE_ASYNC_LDS 0
#endif

__device__ __forceinline__ void async_copy8(const _Float16* g, _Float16* l) {
#if HAVE_ASYNC_LDS
  __builtin_amdgcn_global_load_async_to_lds_b128((gv4ip)(void*)g, (lv4ip)(void*)l, 0, 0);
#else
  *(half8*)l = *(const half8*)g;
#endif
}
__device__ __forceinline__ void async_wait() {
#if HAVE_ASYNC_LDS
#  if __has_builtin(__builtin_amdgcn_s_wait_asynccnt)
  __builtin_amdgcn_s_wait_asynccnt(0);
#  else
  asm volatile("s_wait_asynccnt 0" ::: "memory");
#  endif
#endif
}

__device__ __forceinline__ float sigf(float x) { return 1.f / (1.f + __expf(-x)); }

// ---- WMMA fragment loaders (CDNA5 16x16x32 f16 layouts, wave32) ----
// A (16x32, MxK): lane M = lane&15; halves j -> K = kb + j (kb = 0 or 16)
__device__ __forceinline__ v16h load_a_frag(const _Float16* arow, int kloc, int lane) {
  int kb = (lane & 16) ? 16 : 0;
  half8 lo = *(const half8*)(arow + kloc + kb);
  half8 hi = *(const half8*)(arow + kloc + kb + 8);
  return __builtin_shufflevector(lo, hi, 0,1,2,3,4,5,6,7,8,9,10,11,12,13,14,15);
}
// B (32x16, KxN): lane N = lane&15; halves j -> K = kb2 + j (one contiguous 32B load)
__device__ __forceinline__ v16h load_b_frag(const _Float16* __restrict__ brow, int kloc, int lane) {
  int kb2 = (lane & 16) ? 16 : 0;
  return *(const v16h*)(brow + kloc + kb2);
}

// ---------------- setup kernels ----------------
__global__ void k_pack_w2(const float* __restrict__ W_ih, const float* __restrict__ W_hh,
                          const float* __restrict__ W_x,  const float* __restrict__ W_hc,
                          _Float16* __restrict__ W2) {
  int idx = blockIdx.x * blockDim.x + threadIdx.x;
  if (idx >= NG * KIN) return;
  int j = idx / KIN, col = idx % KIN;
  float v;
  if (j < 4 * H_) v = (col < 2 * E_) ? W_ih[(size_t)j * (2 * E_) + col]
                                     : W_hh[(size_t)j * H_ + (col - 2 * E_)];
  else {
    int j2 = j - 4 * H_;
    v = (col < 2 * E_) ? W_x[(size_t)j2 * (2 * E_) + col]
                       : W_hc[(size_t)j2 * H_ + (col - 2 * E_)];
  }
  W2[idx] = (_Float16)v;
}

__global__ void k_half_wmlp(const float* __restrict__ Wm, _Float16* __restrict__ Wmh) {
  int idx = blockIdx.x * blockDim.x + threadIdx.x;
  if (idx < VOC * H_) Wmh[idx] = (_Float16)Wm[idx];
}

__global__ void k_build_x(const int* __restrict__ cap, const float* __restrict__ emb,
                          const float* __restrict__ vg, _Float16* __restrict__ X) {
  int idx = blockIdx.x * blockDim.x + threadIdx.x;
  if (idx >= B_ * T_ * 2 * E_) return;
  int b = idx / (T_ * 2 * E_);
  int r = idx % (T_ * 2 * E_);
  int t = r / (2 * E_);
  int e = r % (2 * E_);
  float v = (e < E_) ? emb[(size_t)cap[b * T_ + t] * E_ + e] : vg[(size_t)b * E_ + (e - E_)];
  X[idx] = (_Float16)v;
}

__global__ void k_wvv(const float* __restrict__ V, const float* __restrict__ Wv,
                      float* __restrict__ WvV) {
  int idx = blockIdx.x * blockDim.x + threadIdx.x;
  if (idx >= B_ * KATT * KATT) return;
  int b = idx / (KATT * KATT);
  int r = idx % (KATT * KATT);
  int k = r / KATT, a = r % KATT;
  const float* vrow = V + ((size_t)b * KATT + k) * H_;
  const float* wrow = Wv + (size_t)a * H_;
  float s = 0.f;
  for (int h = 0; h < H_; ++h) s += vrow[h] * wrow[h];
  WvV[idx] = s;
}

__global__ void k_init_state(_Float16* __restrict__ hf, float* __restrict__ c) {
  int idx = blockIdx.x * blockDim.x + threadIdx.x;
  if (idx < B_ * H_) { hf[idx] = (_Float16)0.f; c[idx] = 0.f; }
}

// ---------------- per-step kernels ----------------
// S1: G(128 x 2560) = [x_t || h] (128x1536 f16) x W2cat^T.
// Block: 8 waves; A tile (16x1536, 48KB) staged once in LDS; each wave does 4 N-tiles.
__global__ void k_gates_gemm(const _Float16* __restrict__ X, const _Float16* __restrict__ hf,
                             const _Float16* __restrict__ W2, float* __restrict__ G, int t) {
  __shared__ _Float16 As[16 * KIN];                // 48 KB
  int tid = threadIdx.x;
  int wave = tid >> 5, lane = tid & 31;
  int tm = blockIdx.y;
  // --- stage A tile: rows tm*16..+15, k: [0,1024) from X(:,t,:), [1024,1536) from h ---
  for (int ch = tid; ch < (16 * KIN) / 8; ch += blockDim.x) {
    int row = ch / (KIN / 8);
    int ko  = (ch % (KIN / 8)) * 8;
    int grow = tm * 16 + row;
    const _Float16* src = (ko < 2 * E_)
        ? X + ((size_t)grow * T_ + t) * (2 * E_) + ko
        : hf + (size_t)grow * H_ + (ko - 2 * E_);
    async_copy8(src, As + (size_t)row * KIN + ko);
  }
  async_wait();
  __syncthreads();

  int tn0 = (blockIdx.x * NW + wave) * NT;         // 4 consecutive N-tiles per wave
  const _Float16* arow = As + (size_t)(lane & 15) * KIN;
  const _Float16* brow[NT];
  #pragma unroll
  for (int j = 0; j < NT; ++j)
    brow[j] = W2 + (size_t)((tn0 + j) * 16 + (lane & 15)) * KIN;

  v8f acc[NT] = {};
  for (int ks = 0; ks < KIN; ks += 32) {
    v16h a = load_a_frag(arow, ks, lane);          // ds_load from LDS, shared by 4 WMMAs
    #pragma unroll
    for (int j = 0; j < NT; ++j) {
      v16h b = load_b_frag(brow[j], ks, lane);
      acc[j] = __builtin_amdgcn_wmma_f32_16x16x32_f16(false, a, false, b, (short)0, acc[j], false, false);
    }
  }
  int n = lane & 15, moff = (lane >> 4) * 8;
  #pragma unroll
  for (int j = 0; j < NT; ++j)
    #pragma unroll
    for (int r = 0; r < 8; ++r)
      G[(size_t)(tm * 16 + moff + r) * NG + (tn0 + j) * 16 + n] = acc[j][r];
}

// S2: LSTM pointwise + adaptive attention; one block per batch row
__global__ void k_pointwise(const float* __restrict__ G, const float* __restrict__ b_ih,
                            const float* __restrict__ b_hh, float* __restrict__ c,
                            const float* __restrict__ WvV, const float* __restrict__ W_g,
                            const float* __restrict__ W_h_att, const float* __restrict__ V,
                            _Float16* __restrict__ hf, _Float16* __restrict__ uf) {
  __shared__ float hn[H_], st[H_], gh[KATT], zz[KATT];
  __shared__ float beta_s;
  int b = blockIdx.x, tid = threadIdx.x;
  const float* g_row = G + (size_t)b * NG;
  for (int h = tid; h < H_; h += blockDim.x) {
    float gi = g_row[h]            + b_ih[h]            + b_hh[h];
    float gf = g_row[H_ + h]       + b_ih[H_ + h]       + b_hh[H_ + h];
    float gg = g_row[2 * H_ + h]   + b_ih[2 * H_ + h]   + b_hh[2 * H_ + h];
    float go = g_row[3 * H_ + h]   + b_ih[3 * H_ + h]   + b_hh[3 * H_ + h];
    float cn = sigf(gf) * c[(size_t)b * H_ + h] + sigf(gi) * tanhf(gg);
    float hv = sigf(go) * tanhf(cn);
    float sv = sigf(g_row[4 * H_ + h]) * tanhf(cn);   // sentinel gate: no bias
    c[(size_t)b * H_ + h] = cn;
    hn[h] = hv; st[h] = sv;
  }
  __syncthreads();
  if (tid < KATT) {
    const float* wg = W_g + (size_t)tid * H_;
    float s = 0.f;
    for (int h = 0; h < H_; ++h) s += hn[h] * wg[h];
    gh[tid] = s;
  }
  __syncthreads();
  if (tid < KATT) {
    const float* wv = WvV + ((size_t)b * KATT + tid) * KATT;
    float s = 0.f;
    for (int a = 0; a < KATT; ++a) s += tanhf(wv[a] + gh[a]) * W_h_att[a];
    zz[tid] = s;
  }
  __syncthreads();
  if (tid == 0) {
    float m = zz[0];
    for (int k = 1; k < KATT; ++k) m = fmaxf(m, zz[k]);
    float s = 0.f;
    for (int k = 0; k < KATT; ++k) { zz[k] = __expf(zz[k] - m); s += zz[k]; }
    float inv = 1.f / s;
    for (int k = 0; k < KATT; ++k) zz[k] *= inv;
    beta_s = zz[KATT - 1];
  }
  __syncthreads();
  float beta = beta_s;
  for (int h = tid; h < H_; h += blockDim.x) {
    const float* vb = V + (size_t)b * KATT * H_ + h;
    float ct = 0.f;
    for (int k = 0; k < KATT; ++k) ct += zz[k] * vb[(size_t)k * H_];
    float u = beta * st[h] + (1.f - beta) * ct + hn[h];   // hat_c + h_new
    uf[(size_t)b * H_ + h] = (_Float16)u;
    hf[(size_t)b * H_ + h] = (_Float16)hn[h];
  }
}

// S3: out[:, t, :] = u (128x512 f16) x W_mlp^T + b_mlp.
// Block: 8 waves; A tile (16x512, 16KB) staged in LDS; each wave does 4 N-tiles (guarded, 625 total).
__global__ void k_out_gemm(const _Float16* __restrict__ uf, const _Float16* __restrict__ Wm,
                           const float* __restrict__ bm, float* __restrict__ out, int t) {
  __shared__ _Float16 As[16 * H_];                 // 16 KB
  int tid = threadIdx.x;
  int wave = tid >> 5, lane = tid & 31;
  int tm = blockIdx.y;
  const _Float16* srcA = uf + (size_t)tm * 16 * H_;   // 16 contiguous rows
  for (int ch = tid; ch < (16 * H_) / 8; ch += blockDim.x)
    async_copy8(srcA + (size_t)ch * 8, As + (size_t)ch * 8);
  async_wait();
  __syncthreads();

  int tn0 = (blockIdx.x * NW + wave) * NT;
  const _Float16* arow = As + (size_t)(lane & 15) * H_;
  const _Float16* brow[NT];
  #pragma unroll
  for (int j = 0; j < NT; ++j) {
    int tn = tn0 + j;
    brow[j] = Wm + (size_t)((tn < VOC / 16 ? tn : 0) * 16 + (lane & 15)) * H_;
  }
  v8f acc[NT] = {};
  for (int ks = 0; ks < H_; ks += 32) {
    v16h a = load_a_frag(arow, ks, lane);
    #pragma unroll
    for (int j = 0; j < NT; ++j) {
      if (ks + 64 < H_) __builtin_prefetch(brow[j] + ks + 64, 0, 1);   // global_prefetch_b8
      v16h b = load_b_frag(brow[j], ks, lane);
      acc[j] = __builtin_amdgcn_wmma_f32_16x16x32_f16(false, a, false, b, (short)0, acc[j], false, false);
    }
  }
  int n = lane & 15, moff = (lane >> 4) * 8;
  #pragma unroll
  for (int j = 0; j < NT; ++j) {
    int tn = tn0 + j;
    if (tn >= VOC / 16) continue;
    int vcol = tn * 16 + n;
    float bb = bm[vcol];
    #pragma unroll
    for (int r = 0; r < 8; ++r) {
      int row = tm * 16 + moff + r;                // batch index
      out[((size_t)row * T_ + t) * VOC + vcol] = acc[j][r] + bb;
    }
  }
}

// ---------------- host launcher ----------------
extern "C" void kernel_launch(void* const* d_in, const int* in_sizes, int n_in,
                              void* d_out, int out_size, void* d_ws, size_t ws_size,
                              hipStream_t stream) {
  const float* V     = (const float*)d_in[0];
  const float* v_g   = (const float*)d_in[1];
  const int*   cap   = (const int*)  d_in[2];
  const float* emb   = (const float*)d_in[3];
  const float* W_ih  = (const float*)d_in[4];
  const float* W_hh  = (const float*)d_in[5];
  const float* b_ih  = (const float*)d_in[6];
  const float* b_hh  = (const float*)d_in[7];
  const float* W_x   = (const float*)d_in[8];
  const float* W_hc  = (const float*)d_in[9];
  const float* W_v   = (const float*)d_in[10];
  const float* W_g   = (const float*)d_in[11];
  const float* W_ha  = (const float*)d_in[12];
  // d_in[13] = W_s : unused by the reference
  const float* W_mlp = (const float*)d_in[14];
  const float* b_mlp = (const float*)d_in[15];
  float* out = (float*)d_out;

  char* ws = (char*)d_ws;
  size_t off = 0;
  auto carve = [&](size_t bytes) { void* p = ws + off; off = (off + bytes + 255) & ~(size_t)255; return p; };
  _Float16* W2   = (_Float16*)carve((size_t)NG * KIN * 2);
  _Float16* Wmh  = (_Float16*)carve((size_t)VOC * H_ * 2);
  _Float16* X    = (_Float16*)carve((size_t)B_ * T_ * 2 * E_ * 2);
  float*    WvV  = (float*)   carve((size_t)B_ * KATT * KATT * 4);
  _Float16* hf   = (_Float16*)carve((size_t)B_ * H_ * 2);
  float*    c    = (float*)   carve((size_t)B_ * H_ * 4);
  float*    G    = (float*)   carve((size_t)B_ * NG * 4);
  _Float16* uf   = (_Float16*)carve((size_t)B_ * H_ * 2);
  (void)ws_size; (void)in_sizes; (void)n_in; (void)out_size;

  const int TB = 256;
  k_pack_w2   <<<(NG * KIN + TB - 1) / TB, TB, 0, stream>>>(W_ih, W_hh, W_x, W_hc, W2);
  k_half_wmlp <<<(VOC * H_ + TB - 1) / TB, TB, 0, stream>>>(W_mlp, Wmh);
  k_build_x   <<<(B_ * T_ * 2 * E_ + TB - 1) / TB, TB, 0, stream>>>(cap, emb, v_g, X);
  k_wvv       <<<(B_ * KATT * KATT + TB - 1) / TB, TB, 0, stream>>>(V, W_v, WvV);
  k_init_state<<<(B_ * H_ + TB - 1) / TB, TB, 0, stream>>>(hf, c);

  dim3 g1((NG / 16) / (NW * NT), B_ / 16);                     // 5 x 8 blocks
  dim3 g3((VOC / 16 + NW * NT - 1) / (NW * NT), B_ / 16);      // 20 x 8 blocks
  for (int t = 0; t < T_; ++t) {
    k_gates_gemm<<<g1, 32 * NW, 0, stream>>>(X, hf, W2, G, t);
    k_pointwise <<<B_, 256, 0, stream>>>(G, b_ih, b_hh, c, WvV, W_g, W_ha, V, hf, uf);
    k_out_gemm  <<<g3, 32 * NW, 0, stream>>>(uf, Wmh, b_mlp, out, t);
  }
}
